// RGCN_26173530702075
// MI455X (gfx1250) — matrix operand order repaired and checked by reading, not compiled, not run on hardware
//
#include <hip/hip_runtime.h>
#include <hip/hip_bf16.h>

// RGCN on MI455X (gfx1250), compile-only target.
//
// Strategy (see analysis): dense GEMMs via v_wmma_f32_16x16x32_bf16 with a
// bf16 hi/lo split (3 WMMAs per K-step) for ~fp32 accuracy; D^{-1/2} source
// normalization fused into the A-fragment load; segment-sum via float
// atomics (L2-resident); weights pre-split to transposed bf16 hi/lo tables
// so a full B fragment is one aligned 32-byte load.

#define NJOB 50000
#define NRES 50000
#define NEDGE 300000

typedef __bf16 bf16_t;
typedef __attribute__((ext_vector_type(16))) __bf16 v16bf;
typedef __attribute__((ext_vector_type(8)))  float  v8f;

// ---------------------------------------------------------------- utilities
__global__ __launch_bounds__(256) void zero_kernel(float* __restrict__ p, long long n) {
  long long t = (long long)blockIdx.x * 256 + threadIdx.x;
  if (t < n) p[t] = 0.0f;
}

__global__ __launch_bounds__(256) void degree_kernel(const int* __restrict__ idx,
                                                     float* __restrict__ deg, int e) {
  int t = blockIdx.x * 256 + threadIdx.x;
  if (t < e) atomicAdd(&deg[idx[t]], 1.0f);
}

__global__ __launch_bounds__(256) void rsqrt_kernel(float* __restrict__ p, int n) {
  int t = blockIdx.x * 256 + threadIdx.x;
  if (t < n) p[t] = rsqrtf(fmaxf(p[t], 1.0f));
}

// Split fp32 W[K][N] into transposed bf16 hi/lo tables WT[N][K].
__global__ __launch_bounds__(256) void split_w_kernel(const float* __restrict__ W,
                                                      bf16_t* __restrict__ hiT,
                                                      bf16_t* __restrict__ loT,
                                                      int K, int N) {
  int t = blockIdx.x * 256 + threadIdx.x;
  if (t >= K * N) return;
  int k = t / N, n = t - k * N;
  float w = W[t];
  bf16_t h = (bf16_t)w;
  hiT[(size_t)n * K + k] = h;
  loT[(size_t)n * K + k] = (bf16_t)(w - (float)h);
}

// ------------------------------------------------------------- WMMA GEMM
// C[M][N] = diag(nsrc) * A[M][K] @ W[K][N], bf16x3 split, f32 accumulate.
// Block: 256 thr = 8 waves; wave w -> rows [bx*128 + 16w, +16), cols [by*64, +64).
__global__ __launch_bounds__(256)
void gemm_wmma_kernel(const float* __restrict__ A, const float* __restrict__ nsrc,
                      const bf16_t* __restrict__ WhiT, const bf16_t* __restrict__ WloT,
                      float* __restrict__ C, int M, int K, int N) {
  const int lane = threadIdx.x & 31;
  const int wave = threadIdx.x >> 5;
  const int half = lane >> 4;    // lane group 0-15 / 16-31
  const int ln   = lane & 15;
  const int rowBase = blockIdx.x * 128 + wave * 16;
  const int colBase = blockIdx.y * 64;

  const int m  = rowBase + ln;                 // A row this lane feeds
  const int mc = (m < M) ? m : (M - 1);        // clamp tail (stores predicated)
  const float scale = nsrc[mc];
  const float* __restrict__ arow = A + (size_t)mc * K;

  v8f zero8 = {};
  v8f acc[4];
#pragma unroll
  for (int i = 0; i < 4; ++i) acc[i] = zero8;

  for (int kk = 0; kk < K; kk += 32) {
    // A fragment element e: K = kk + 16*half + e
    const int k0 = kk + 16 * half;
    const float4* __restrict__ pa = (const float4*)(arow + k0);
    const float4* __restrict__ pb = (const float4*)(arow + k0 + 8);
    float4 q0 = pa[0], q1 = pa[1], q2 = pb[0], q3 = pb[1];
    float af[16];
    af[0]=q0.x; af[1]=q0.y; af[2]=q0.z; af[3]=q0.w;
    af[4]=q1.x; af[5]=q1.y; af[6]=q1.z; af[7]=q1.w;
    af[8]=q2.x; af[9]=q2.y; af[10]=q2.z; af[11]=q2.w;
    af[12]=q3.x; af[13]=q3.y; af[14]=q3.z; af[15]=q3.w;

    v16bf ahi, alo;
#pragma unroll
    for (int e = 0; e < 16; ++e) {
      float v = af[e] * scale;
      bf16_t h = (bf16_t)v;
      ahi[e] = h;
      alo[e] = (bf16_t)(v - (float)h);
    }

    // B fragment element e: K = kk + 16*half + e (contiguous in transposed W)
    const int kb = kk + 16 * half;
#pragma unroll
    for (int nt = 0; nt < 4; ++nt) {
      const int col = colBase + nt * 16 + ln;
      const v16bf bhi = *(const v16bf*)(WhiT + (size_t)col * K + kb);
      const v16bf blo = *(const v16bf*)(WloT + (size_t)col * K + kb);
      acc[nt] = __builtin_amdgcn_wmma_f32_16x16x32_bf16(false, ahi, false, bhi,
                                                        (short)0, acc[nt], false, false);
      acc[nt] = __builtin_amdgcn_wmma_f32_16x16x32_bf16(false, alo, false, bhi,
                                                        (short)0, acc[nt], false, false);
      acc[nt] = __builtin_amdgcn_wmma_f32_16x16x32_bf16(false, ahi, false, blo,
                                                        (short)0, acc[nt], false, false);
    }
  }

  // C/D layout: VGPR r -> row (r + 8*half), col = ln
#pragma unroll
  for (int nt = 0; nt < 4; ++nt) {
    const int col = colBase + nt * 16 + ln;
#pragma unroll
    for (int r = 0; r < 8; ++r) {
      const int row = rowBase + r + 8 * half;
      if (row < M) C[(size_t)row * N + col] = acc[nt][r];
    }
  }
}

// ------------------------------------------------------- edge scatter-add
// agg[dst[e]][:] += h[src[e]][:]; D4 = D/4; one block covers 256/D4 edges.
__global__ __launch_bounds__(256)
void scatter_add_kernel(const float* __restrict__ h, const int* __restrict__ src,
                        const int* __restrict__ dst, float* __restrict__ agg,
                        int ne, int D4) {
  const int per = 256 / D4;
  const int le  = threadIdx.x / D4;
  const int c4  = threadIdx.x - le * D4;
  const int e   = blockIdx.x * per + le;
  if (e >= ne) return;
  const int s = src[e];
  const int d = dst[e];
  const float4 v = ((const float4*)h)[(size_t)s * D4 + c4];
  float* p = agg + (((size_t)d * D4 + c4) << 2);
  atomicAdd(p + 0, v.x);
  atomicAdd(p + 1, v.y);
  atomicAdd(p + 2, v.z);
  atomicAdd(p + 3, v.w);
}

// ------------------------------------------------------------ finalize
__global__ __launch_bounds__(256)
void finalize_job_kernel(const float* __restrict__ agg_a, const float* __restrict__ agg_s,
                         const float* __restrict__ nd_a, const float* __restrict__ nd_s,
                         const float* __restrict__ ba, const float* __restrict__ bs,
                         float* __restrict__ out, int n, int dshift, int do_relu) {
  const long long total = (long long)n << dshift;
  long long t = (long long)blockIdx.x * 256 + threadIdx.x;
  if (t >= total) return;
  const int i = (int)(t >> dshift);
  const int d = (int)(t & ((1 << dshift) - 1));
  float v = fmaf(agg_a[t], nd_a[i], ba[d]) + fmaf(agg_s[t], nd_s[i], bs[d]);
  out[t] = do_relu ? fmaxf(v, 0.0f) : v;
}

__global__ __launch_bounds__(256)
void finalize_res_kernel(const float* __restrict__ agg_m, const float* __restrict__ nd_m,
                         const float* __restrict__ bm, float* __restrict__ out,
                         int n, int dshift, int do_relu) {
  const long long total = (long long)n << dshift;
  long long t = (long long)blockIdx.x * 256 + threadIdx.x;
  if (t >= total) return;
  const int i = (int)(t >> dshift);
  const int d = (int)(t & ((1 << dshift) - 1));
  float v = fmaf(agg_m[t], nd_m[i], bm[d]);
  out[t] = do_relu ? fmaxf(v, 0.0f) : v;
}

// ---------------------------------------------------------------- driver
extern "C" void kernel_launch(void* const* d_in, const int* in_sizes, int n_in,
                              void* d_out, int out_size, void* d_ws, size_t ws_size,
                              hipStream_t stream) {
  (void)in_sizes; (void)n_in; (void)out_size; (void)ws_size;
  const int NJ = NJOB, NR = NRES, E = NEDGE;

  const float* x_job = (const float*)d_in[0];
  const float* x_res = (const float*)d_in[1];
  const int* srcA = (const int*)d_in[2];   // applies: res -> job
  const int* dstA = (const int*)d_in[3];
  const int* srcM = (const int*)d_in[4];   // matches: job -> res
  const int* dstM = (const int*)d_in[5];
  const int* srcS = (const int*)d_in[6];   // similar: job -> job
  const int* dstS = (const int*)d_in[7];

  // dict order: per layer, per rel in (a, m, s): W then b
  const float *W[3][3], *Bv[3][3];
  {
    int p = 8;
    for (int L = 0; L < 3; ++L)
      for (int r = 0; r < 3; ++r) { W[L][r] = (const float*)d_in[p++]; Bv[L][r] = (const float*)d_in[p++]; }
  }

  // workspace carve-up (floats); ~309 MB total
  float* ws = (float*)d_ws;
  size_t off = 0;
  auto alloc = [&](size_t nf) { float* p = ws + off; off += nf; return p; };
  float* normbase = ws;                 // 6 contiguous degree->norm arrays
  float* ns_a = alloc(NR);  float* nd_a = alloc(NJ);
  float* ns_s = alloc(NJ);  float* nd_s = alloc(NJ);
  float* ns_m = alloc(NJ);  float* nd_m = alloc(NR);
  const long long NNORM = (long long)off;            // 300000
  float* bufJ  = alloc((size_t)NJ * 256);            // ping-pong features (in-place across layers)
  float* bufR  = alloc((size_t)NR * 256);
  float* h_tmp = alloc((size_t)50000 * 256);
  float* agg_a = alloc((size_t)NJ * 256);
  float* agg_s = alloc((size_t)NJ * 256);
  float* agg_m = alloc((size_t)NR * 256);
  off = (off + 15) & ~(size_t)15;                    // 64B align for v16bf loads
  bf16_t* wsp = (bf16_t*)(ws + off);                 // 3 rels x (hi+lo) x 256*256 bf16
  bf16_t *whiT[3], *wloT[3];
  for (int r = 0; r < 3; ++r) { whiT[r] = wsp + (size_t)r * 2 * 65536; wloT[r] = whiT[r] + 65536; }

  dim3 blk(256);
  auto cdiv = [](long long a, long long b) { return (unsigned)((a + b - 1) / b); };

  // ---- degrees -> norms (edges are layer-invariant; recomputed each call)
  zero_kernel<<<cdiv(NNORM, 256), blk, 0, stream>>>(normbase, NNORM);
  degree_kernel<<<cdiv(E, 256), blk, 0, stream>>>(srcA, ns_a, E);
  degree_kernel<<<cdiv(E, 256), blk, 0, stream>>>(dstA, nd_a, E);
  degree_kernel<<<cdiv(E, 256), blk, 0, stream>>>(srcS, ns_s, E);
  degree_kernel<<<cdiv(E, 256), blk, 0, stream>>>(dstS, nd_s, E);
  degree_kernel<<<cdiv(E, 256), blk, 0, stream>>>(srcM, ns_m, E);
  degree_kernel<<<cdiv(E, 256), blk, 0, stream>>>(dstM, nd_m, E);
  rsqrt_kernel<<<cdiv(NNORM, 256), blk, 0, stream>>>(normbase, (int)NNORM);

  const float* curJ = x_job;
  const float* curR = x_res;
  float* outJ[3] = {bufJ, bufJ, (float*)d_out};
  float* outR[3] = {bufR, bufR, (float*)d_out + (size_t)NJ * 128};

  for (int L = 0; L < 3; ++L) {
    const int din  = 256;
    const int dout = (L == 2) ? 128 : 256;
    const int D4   = dout / 4;
    const int per  = 256 / D4;
    const int dshift = (dout == 256) ? 8 : 7;

    for (int r = 0; r < 3; ++r)
      split_w_kernel<<<cdiv((long long)din * dout, 256), blk, 0, stream>>>(
          W[L][r], whiT[r], wloT[r], din, dout);

    zero_kernel<<<cdiv((long long)NJ * dout, 256), blk, 0, stream>>>(agg_a, (long long)NJ * dout);
    zero_kernel<<<cdiv((long long)NJ * dout, 256), blk, 0, stream>>>(agg_s, (long long)NJ * dout);
    zero_kernel<<<cdiv((long long)NR * dout, 256), blk, 0, stream>>>(agg_m, (long long)NR * dout);

    // applies: h = (x_res * ns_a) @ Wa ; scatter res->job
    gemm_wmma_kernel<<<dim3(cdiv(NR, 128), dout / 64), blk, 0, stream>>>(
        curR, ns_a, whiT[0], wloT[0], h_tmp, NR, din, dout);
    scatter_add_kernel<<<cdiv(E, per), blk, 0, stream>>>(h_tmp, srcA, dstA, agg_a, E, D4);

    // similar: h = (x_job * ns_s) @ Ws ; scatter job->job
    gemm_wmma_kernel<<<dim3(cdiv(NJ, 128), dout / 64), blk, 0, stream>>>(
        curJ, ns_s, whiT[2], wloT[2], h_tmp, NJ, din, dout);
    scatter_add_kernel<<<cdiv(E, per), blk, 0, stream>>>(h_tmp, srcS, dstS, agg_s, E, D4);

    // matches: h = (x_job * ns_m) @ Wm ; scatter job->res
    gemm_wmma_kernel<<<dim3(cdiv(NJ, 128), dout / 64), blk, 0, stream>>>(
        curJ, ns_m, whiT[1], wloT[1], h_tmp, NJ, din, dout);
    scatter_add_kernel<<<cdiv(E, per), blk, 0, stream>>>(h_tmp, srcM, dstM, agg_m, E, D4);

    const int relu = (L < 2) ? 1 : 0;
    finalize_job_kernel<<<cdiv((long long)NJ * dout, 256), blk, 0, stream>>>(
        agg_a, agg_s, nd_a, nd_s, Bv[L][0], Bv[L][2], outJ[L], NJ, dshift, relu);
    finalize_res_kernel<<<cdiv((long long)NR * dout, 256), blk, 0, stream>>>(
        agg_m, nd_m, Bv[L][1], outR[L], NR, dshift, relu);

    curJ = outJ[L];
    curR = outR[L];
  }
}
